// QuantTwoWayAttentionBlock_86835648790566
// MI455X (gfx1250) — 8-bit matrix operand order repaired and checked by reading, compiled, not run on hardware
//
#include <hip/hip_runtime.h>
#include <stdint.h>

#define BB   32
#define NQ_  8
#define NK_  4096
#define E_   256
#define H_   8
#define MLP_ 2048
#define CH2  128
#define TQ   (BB * NQ_)
#define TK   (BB * NK_)

typedef __attribute__((ext_vector_type(8))) int v8i;

__device__ __forceinline__ float scale_of(float amax) {
  return fmaxf(amax * (1.0f / 127.0f), 1e-8f);
}
__device__ __forceinline__ float fqv(float x, float s) {
  float r = rintf(x / s);
  r = fminf(fmaxf(r, -128.0f), 127.0f);
  return r * s;
}

__global__ __launch_bounds__(256) void k_zero(float* p, int n) {
  int i = blockIdx.x * 256 + threadIdx.x;
  if (i < n) p[i] = 0.0f;
}

__global__ __launch_bounds__(256) void k_absmax(const float* __restrict__ x,
                                                long long n, float* slot) {
  __shared__ float red[256];
  float m = 0.0f;
  for (long long i = (long long)blockIdx.x * 256 + threadIdx.x; i < n;
       i += (long long)gridDim.x * 256)
    m = fmaxf(m, fabsf(x[i]));
  red[threadIdx.x] = m;
  __syncthreads();
  for (int st = 128; st > 0; st >>= 1) {
    if (threadIdx.x < st) red[threadIdx.x] = fmaxf(red[threadIdx.x], red[threadIdx.x + st]);
    __syncthreads();
  }
  if (threadIdx.x == 0) atomicMax((unsigned int*)slot, __float_as_uint(red[0]));
}

__global__ __launch_bounds__(256) void k_absmax2(const float* __restrict__ a,
                                                 const float* __restrict__ b,
                                                 long long n, float* slot) {
  __shared__ float red[256];
  float m = 0.0f;
  for (long long i = (long long)blockIdx.x * 256 + threadIdx.x; i < n;
       i += (long long)gridDim.x * 256)
    m = fmaxf(m, fabsf(a[i] + b[i]));
  red[threadIdx.x] = m;
  __syncthreads();
  for (int st = 128; st > 0; st >>= 1) {
    if (threadIdx.x < st) red[threadIdx.x] = fmaxf(red[threadIdx.x], red[threadIdx.x + st]);
    __syncthreads();
  }
  if (threadIdx.x == 0) atomicMax((unsigned int*)slot, __float_as_uint(red[0]));
}

__global__ __launch_bounds__(256) void k_quant(const float* __restrict__ x,
                                               const float* __restrict__ amax,
                                               signed char* __restrict__ q, long long n) {
  long long i = (long long)blockIdx.x * 256 + threadIdx.x;
  if (i >= n) return;
  float s = scale_of(amax[0]);
  float r = rintf(x[i] / s);
  r = fminf(fmaxf(r, -128.0f), 127.0f);
  q[i] = (signed char)r;
}

__global__ __launch_bounds__(256) void k_quant2(const float* __restrict__ a,
                                                const float* __restrict__ b,
                                                const float* __restrict__ amax,
                                                signed char* __restrict__ q, long long n) {
  long long i = (long long)blockIdx.x * 256 + threadIdx.x;
  if (i >= n) return;
  float s = scale_of(amax[0]);
  float r = rintf((a[i] + b[i]) / s);
  r = fminf(fmaxf(r, -128.0f), 127.0f);
  q[i] = (signed char)r;
}

__global__ __launch_bounds__(256)
void k_gemm_i8(const signed char* __restrict__ A,
               const signed char* __restrict__ W,
               const float* __restrict__ bias,
               const float* __restrict__ amaxA,
               const float* __restrict__ amaxW,
               const float* residual,
               float* C, int M, int N, int K, int relu) {
  __shared__ signed char smA[128 * 64];
  const int tid  = threadIdx.x;
  const int wave = tid >> 5;
  const int lane = tid & 31;
  const int l16  = lane & 15;
  const int lhi  = lane >> 4;
  const int mBase = blockIdx.x * 128;
  const float scale = scale_of(amaxA[0]) * scale_of(amaxW[0]);

  const int srow = tid >> 1;
  const int scol = (tid & 1) * 32;

  for (int n0 = 0; n0 < N; n0 += 64) {
    v8i acc[4];
#pragma unroll
    for (int t = 0; t < 4; ++t) acc[t] = (v8i){0, 0, 0, 0, 0, 0, 0, 0};

    for (int k0 = 0; k0 < K; k0 += 64) {
      __syncthreads();
      {
        const int4* src = (const int4*)(A + (long long)(mBase + srow) * K + k0 + scol);
        int4* dst = (int4*)(smA + srow * 64 + scol);
        dst[0] = src[0];
        dst[1] = src[1];
        if (k0 + 64 < K)
          __builtin_prefetch(A + (long long)(mBase + srow) * K + k0 + 64 + scol, 0, 2);
      }
      __syncthreads();

      v8i a;
      {
        const signed char* p = smA + (wave * 16 + l16) * 64 + lhi * 16;
        int4 d0 = *(const int4*)(p);
        int4 d1 = *(const int4*)(p + 32);
        a = (v8i){d0.x, d0.y, d0.z, d0.w, d1.x, d1.y, d1.z, d1.w};
      }
#pragma unroll
      for (int t = 0; t < 4; ++t) {
        const signed char* p = W + (long long)(n0 + t * 16 + l16) * K + k0 + lhi * 16;
        int4 b0 = *(const int4*)(p);
        int4 b1 = *(const int4*)(p + 32);
        v8i bf = (v8i){b0.x, b0.y, b0.z, b0.w, b1.x, b1.y, b1.z, b1.w};
        acc[t] = __builtin_amdgcn_wmma_i32_16x16x64_iu8(true, a, true, bf, acc[t],
                                                        false, false);
      }
    }
#pragma unroll
    for (int t = 0; t < 4; ++t) {
      const int col = n0 + t * 16 + l16;
      const float bv = bias ? bias[col] : 0.0f;
#pragma unroll
      for (int r = 0; r < 8; ++r) {
        const long long row = mBase + wave * 16 + lhi * 8 + r;
        float v = scale * (float)acc[t][r] + bv;
        if (residual) v += residual[row * (long long)N + col];
        if (relu) v = fmaxf(v, 0.0f);
        C[row * (long long)N + col] = v;
      }
    }
  }
}

__global__ __launch_bounds__(256)
void k_attn_probs_block(const float* __restrict__ Q, const float* __restrict__ K,
                        const float* __restrict__ amaxQ, const float* __restrict__ amaxK,
                        float* __restrict__ probs, int Hh, int Nq, int Nk, int d) {
  __shared__ float red[256];
  const int idx = blockIdx.x;
  const int nq = idx % Nq;
  const int bh = idx / Nq;
  const int h = bh % Hh;
  const int b = bh / Hh;
  const int CH = Hh * d;
  const float sq = scale_of(amaxQ[0]);
  const float sk = scale_of(amaxK[0]);
  const float inv = rsqrtf((float)d);
  const float* qrow = Q + ((long long)(b * Nq + nq)) * CH + h * d;
  float* prow = probs + (long long)idx * Nk;

  float lmax = -1e30f;
  for (int nk = threadIdx.x; nk < Nk; nk += 256) {
    const float* krow = K + ((long long)(b * Nk + nk)) * CH + h * d;
    float s = 0.0f;
    for (int c = 0; c < d; ++c) s += fqv(qrow[c], sq) * fqv(krow[c], sk);
    s *= inv;
    prow[nk] = s;
    lmax = fmaxf(lmax, s);
  }
  red[threadIdx.x] = lmax;
  __syncthreads();
  for (int st = 128; st > 0; st >>= 1) {
    if (threadIdx.x < st) red[threadIdx.x] = fmaxf(red[threadIdx.x], red[threadIdx.x + st]);
    __syncthreads();
  }
  const float m = red[0];
  __syncthreads();
  float lsum = 0.0f;
  for (int nk = threadIdx.x; nk < Nk; nk += 256) {
    float e = __expf(prow[nk] - m);
    prow[nk] = e;
    lsum += e;
  }
  red[threadIdx.x] = lsum;
  __syncthreads();
  for (int st = 128; st > 0; st >>= 1) {
    if (threadIdx.x < st) red[threadIdx.x] += red[threadIdx.x + st];
    __syncthreads();
  }
  const float is = 1.0f / red[0];
  for (int nk = threadIdx.x; nk < Nk; nk += 256) prow[nk] *= is;
}

__global__ __launch_bounds__(256)
void k_attn_probs_small(const float* __restrict__ Q, const float* __restrict__ K,
                        const float* __restrict__ amaxQ, const float* __restrict__ amaxK,
                        float* __restrict__ probs, int Hh, int Nq, int Nk, int d,
                        long long rows) {
  long long idx = (long long)blockIdx.x * 256 + threadIdx.x;
  if (idx >= rows) return;
  const int nq = (int)(idx % Nq);
  long long bh = idx / Nq;
  const int h = (int)(bh % Hh);
  const int b = (int)(bh / Hh);
  const int CH = Hh * d;
  const float sq = scale_of(amaxQ[0]);
  const float sk = scale_of(amaxK[0]);
  const float inv = rsqrtf((float)d);
  const float* qrow = Q + ((long long)(b * Nq + nq)) * CH + h * d;
  float sc[16];
  float m = -1e30f;
  for (int nk = 0; nk < Nk; ++nk) {
    const float* krow = K + ((long long)(b * Nk + nk)) * CH + h * d;
    float s = 0.0f;
    for (int c = 0; c < d; ++c) s += fqv(qrow[c], sq) * fqv(krow[c], sk);
    s *= inv;
    sc[nk] = s;
    m = fmaxf(m, s);
  }
  float sum = 0.0f;
  for (int nk = 0; nk < Nk; ++nk) {
    float e = __expf(sc[nk] - m);
    sc[nk] = e;
    sum += e;
  }
  const float is = 1.0f / sum;
  float* prow = probs + idx * Nk;
  for (int nk = 0; nk < Nk; ++nk) prow[nk] = sc[nk] * is;
}

__global__ __launch_bounds__(256)
void k_attn_av(const float* __restrict__ probs, const float* __restrict__ V,
               const float* __restrict__ amaxP, const float* __restrict__ amaxV,
               float* __restrict__ out, int Hh, int Nq, int Nk, int d, long long total) {
  long long t = (long long)blockIdx.x * 256 + threadIdx.x;
  if (t >= total) return;
  const int c = (int)(t % d);
  long long rowi = t / d;
  const int nq = (int)(rowi % Nq);
  long long bh = rowi / Nq;
  const int h = (int)(bh % Hh);
  const int b = (int)(bh / Hh);
  const int CH = Hh * d;
  const float sp = scale_of(amaxP[0]);
  const float sv = scale_of(amaxV[0]);
  const float* prow = probs + rowi * Nk;
  float acc = 0.0f;
  for (int nk = 0; nk < Nk; ++nk) {
    float pv = fqv(prow[nk], sp);
    float vv = fqv(V[((long long)(b * Nk + nk)) * CH + h * d + c], sv);
    acc += pv * vv;
  }
  out[((long long)(b * Nq + nq)) * CH + h * d + c] = acc;
}

__global__ __launch_bounds__(256)
void k_layernorm(float* __restrict__ x, const float* __restrict__ w,
                 const float* __restrict__ b) {
  __shared__ float r1[256], r2[256];
  const long long row = blockIdx.x;
  const int t = threadIdx.x;
  float v = x[row * 256 + t];
  r1[t] = v;
  r2[t] = v * v;
  __syncthreads();
  for (int st = 128; st > 0; st >>= 1) {
    if (t < st) { r1[t] += r1[t + st]; r2[t] += r2[t + st]; }
    __syncthreads();
  }
  const float mean = r1[0] * (1.0f / 256.0f);
  const float var = r2[0] * (1.0f / 256.0f) - mean * mean;
  x[row * 256 + t] = (v - mean) * rsqrtf(var + 1e-5f) * w[t] + b[t];
}

extern "C" void kernel_launch(void* const* d_in, const int* in_sizes, int n_in,
                              void* d_out, int out_size, void* d_ws, size_t ws_size,
                              hipStream_t stream) {
  (void)in_sizes; (void)n_in; (void)out_size; (void)ws_size;
  const float* IN_Q   = (const float*)d_in[0];
  const float* IN_K   = (const float*)d_in[1];
  const float* IN_QPE = (const float*)d_in[2];
  const float* IN_KPE = (const float*)d_in[3];
  const float* P[36];
  for (int i = 0; i < 36; ++i) P[i] = (const float*)d_in[4 + i];

  float* QCUR = (float*)d_out;
  float* KOUT = (float*)d_out + (size_t)TQ * E_;

  char* base = (char*)d_ws;
  size_t cur = 0;
  auto alloc = [&](size_t bytes) -> void* {
    void* p = base + cur;
    cur += (bytes + 255) & ~(size_t)255;
    return p;
  };

  float* AM = (float*)alloc(64 * sizeof(float));
  signed char* X8a  = (signed char*)alloc((size_t)TQ * MLP_);
  signed char* X8b  = (signed char*)alloc((size_t)TQ * E_);
  signed char* W8   = (signed char*)alloc((size_t)MLP_ * E_);
  signed char* KPE8 = (signed char*)alloc((size_t)TK * E_);
  signed char* K8   = (signed char*)alloc((size_t)TK * E_);
  signed char* AO8  = (signed char*)alloc((size_t)TK * CH2);
  float* SAQ   = (float*)alloc((size_t)TQ * E_ * 4);
  float* SAK   = (float*)alloc((size_t)TQ * E_ * 4);
  float* SAV   = (float*)alloc((size_t)TQ * E_ * 4);
  float* SAAO  = (float*)alloc((size_t)TQ * E_ * 4);
  float* SAPR  = (float*)alloc((size_t)BB * H_ * NQ_ * NQ_ * 4);
  float* HBUF  = (float*)alloc((size_t)TQ * MLP_ * 4);
  float* T2IQ  = (float*)alloc((size_t)TQ * CH2 * 4);
  float* T2IAO = (float*)alloc((size_t)TQ * CH2 * 4);
  float* QK    = (float*)alloc((size_t)TQ * CH2 * 4);
  float* QV    = (float*)alloc((size_t)TQ * CH2 * 4);
  float* PROBS = (float*)alloc((size_t)BB * H_ * NQ_ * NK_ * 4);
  float* BIGA  = (float*)alloc((size_t)TK * CH2 * 4);
  float* BIGB  = (float*)alloc((size_t)TK * CH2 * 4);

  auto gridOf = [](long long n) { return dim3((unsigned)((n + 255) / 256)); };
  auto absmax1 = [&](const float* x, long long n, int slot) {
    long long g = (n + 255) / 256; if (g > 4096) g = 4096;
    k_absmax<<<dim3((unsigned)g), 256, 0, stream>>>(x, n, AM + slot);
  };
  auto absmax2f = [&](const float* a, const float* b, long long n, int slot) {
    long long g = (n + 255) / 256; if (g > 4096) g = 4096;
    k_absmax2<<<dim3((unsigned)g), 256, 0, stream>>>(a, b, n, AM + slot);
  };
  auto quant1 = [&](const float* x, int slot, signed char* q, long long n) {
    k_quant<<<gridOf(n), 256, 0, stream>>>(x, AM + slot, q, n);
  };
  auto quant2f = [&](const float* a, const float* b, int slot, signed char* q, long long n) {
    k_quant2<<<gridOf(n), 256, 0, stream>>>(a, b, AM + slot, q, n);
  };
  auto qweight = [&](const float* Wf, int slot, int N, int K) {
    absmax1(Wf, (long long)N * K, slot);
    quant1(Wf, slot, W8, (long long)N * K);
  };
  auto gemm = [&](const signed char* A8, int sA, int sW, const float* bias,
                  const float* resid, float* Cp, int M, int N, int K, int relu) {
    k_gemm_i8<<<dim3(M / 128), 256, 0, stream>>>(A8, W8, bias, AM + sA, AM + sW,
                                                 resid, Cp, M, N, K, relu);
  };
  auto lnorm = [&](float* x, const float* w, const float* b, long long T) {
    k_layernorm<<<dim3((unsigned)T), 256, 0, stream>>>(x, w, b);
  };

  k_zero<<<1, 256, 0, stream>>>(AM, 64);

  absmax2f(IN_Q, IN_QPE, (long long)TQ * E_, 0);
  quant2f(IN_Q, IN_QPE, 0, X8a, (long long)TQ * E_);
  absmax1(IN_Q, (long long)TQ * E_, 1);
  quant1(IN_Q, 1, X8b, (long long)TQ * E_);
  qweight(P[0], 2, E_, E_);  gemm(X8a, 0, 2, P[1], nullptr, SAQ, TQ, E_, E_, 0);
  qweight(P[2], 3, E_, E_);  gemm(X8a, 0, 3, P[3], nullptr, SAK, TQ, E_, E_, 0);
  qweight(P[4], 4, E_, E_);  gemm(X8b, 1, 4, P[5], nullptr, SAV, TQ, E_, E_, 0);
  absmax1(SAQ, (long long)TQ * E_, 5);
  absmax1(SAK, (long long)TQ * E_, 6);
  absmax1(SAV, (long long)TQ * E_, 7);
  {
    long long rows = (long long)BB * H_ * NQ_;
    k_attn_probs_small<<<gridOf(rows), 256, 0, stream>>>(SAQ, SAK, AM + 5, AM + 6,
                                                         SAPR, H_, NQ_, NQ_, 32, rows);
  }
  absmax1(SAPR, (long long)BB * H_ * NQ_ * NQ_, 8);
  {
    long long tot = (long long)BB * H_ * NQ_ * 32;
    k_attn_av<<<gridOf(tot), 256, 0, stream>>>(SAPR, SAV, AM + 8, AM + 7, SAAO,
                                               H_, NQ_, NQ_, 32, tot);
  }
  absmax1(SAAO, (long long)TQ * E_, 9);
  quant1(SAAO, 9, X8a, (long long)TQ * E_);
  qweight(P[6], 10, E_, E_);
  gemm(X8a, 9, 10, P[7], IN_Q, QCUR, TQ, E_, E_, 0);
  lnorm(QCUR, P[28], P[29], TQ);

  absmax2f(QCUR, IN_QPE, (long long)TQ * E_, 11);
  quant2f(QCUR, IN_QPE, 11, X8a, (long long)TQ * E_);
  absmax2f(IN_K, IN_KPE, (long long)TK * E_, 12);
  quant2f(IN_K, IN_KPE, 12, KPE8, (long long)TK * E_);
  absmax1(IN_K, (long long)TK * E_, 13);
  quant1(IN_K, 13, K8, (long long)TK * E_);
  qweight(P[8],  14, CH2, E_); gemm(X8a,  11, 14, P[9],  nullptr, T2IQ, TQ, CH2, E_, 0);
  qweight(P[10], 15, CH2, E_); gemm(KPE8, 12, 15, P[11], nullptr, BIGA, TK, CH2, E_, 0);
  qweight(P[12], 16, CH2, E_); gemm(K8,   13, 16, P[13], nullptr, BIGB, TK, CH2, E_, 0);
  absmax1(T2IQ, (long long)TQ * CH2, 17);
  absmax1(BIGA, (long long)TK * CH2, 18);
  absmax1(BIGB, (long long)TK * CH2, 19);
  k_attn_probs_block<<<dim3(BB * H_ * NQ_), 256, 0, stream>>>(T2IQ, BIGA, AM + 17,
                                                              AM + 18, PROBS, H_, NQ_,
                                                              NK_, 16);
  absmax1(PROBS, (long long)BB * H_ * NQ_ * NK_, 20);
  {
    long long tot = (long long)BB * H_ * NQ_ * 16;
    k_attn_av<<<gridOf(tot), 256, 0, stream>>>(PROBS, BIGB, AM + 20, AM + 19, T2IAO,
                                               H_, NQ_, NK_, 16, tot);
  }
  absmax1(T2IAO, (long long)TQ * CH2, 21);
  quant1(T2IAO, 21, X8a, (long long)TQ * CH2);
  qweight(P[14], 22, E_, CH2);
  gemm(X8a, 21, 22, P[15], QCUR, QCUR, TQ, E_, CH2, 0);
  lnorm(QCUR, P[30], P[31], TQ);

  absmax1(QCUR, (long long)TQ * E_, 23);
  quant1(QCUR, 23, X8a, (long long)TQ * E_);
  qweight(P[24], 24, MLP_, E_);
  gemm(X8a, 23, 24, P[25], nullptr, HBUF, TQ, MLP_, E_, 1);
  absmax1(HBUF, (long long)TQ * MLP_, 25);
  quant1(HBUF, 25, X8a, (long long)TQ * MLP_);
  qweight(P[26], 26, E_, MLP_);
  gemm(X8a, 25, 26, P[27], QCUR, QCUR, TQ, E_, MLP_, 0);
  lnorm(QCUR, P[32], P[33], TQ);

  absmax2f(QCUR, IN_QPE, (long long)TQ * E_, 27);
  quant2f(QCUR, IN_QPE, 27, X8a, (long long)TQ * E_);
  qweight(P[16], 28, CH2, E_); gemm(KPE8, 12, 28, P[17], nullptr, BIGA, TK, CH2, E_, 0);
  qweight(P[18], 29, CH2, E_); gemm(X8a,  27, 29, P[19], nullptr, QK,   TQ, CH2, E_, 0);
  absmax1(QCUR, (long long)TQ * E_, 30);
  quant1(QCUR, 30, X8b, (long long)TQ * E_);
  qweight(P[20], 31, CH2, E_); gemm(X8b,  30, 31, P[21], nullptr, QV,   TQ, CH2, E_, 0);
  absmax1(BIGA, (long long)TK * CH2, 32);
  absmax1(QK, (long long)TQ * CH2, 33);
  absmax1(QV, (long long)TQ * CH2, 34);
  {
    long long rows = (long long)BB * H_ * NK_;
    k_attn_probs_small<<<gridOf(rows), 256, 0, stream>>>(BIGA, QK, AM + 32, AM + 33,
                                                         PROBS, H_, NK_, NQ_, 16, rows);
  }
  absmax1(PROBS, (long long)BB * H_ * NK_ * NQ_, 35);
  {
    long long tot = (long long)BB * H_ * NK_ * 16;
    k_attn_av<<<gridOf(tot), 256, 0, stream>>>(PROBS, QV, AM + 35, AM + 34, BIGB,
                                               H_, NK_, NQ_, 16, tot);
  }
  absmax1(BIGB, (long long)TK * CH2, 36);
  quant1(BIGB, 36, AO8, (long long)TK * CH2);
  qweight(P[22], 37, E_, CH2);
  gemm(AO8, 36, 37, P[23], IN_K, KOUT, TK, E_, CH2, 0);
  lnorm(KOUT, P[34], P[35], TK);
}
